// LSTMdecoder_58093727645744
// MI455X (gfx1250) — matrix operand order repaired and checked by reading, not compiled, not run on hardware
//
#include <hip/hip_runtime.h>

typedef __bf16 bf16;
typedef __attribute__((ext_vector_type(16))) __bf16 v16bf;
typedef __attribute__((ext_vector_type(8)))  __bf16 v8bf;
typedef __attribute__((ext_vector_type(8)))  float  v8f;

#define T_N 128
#define B_N 64
#define S_N 256
#define K_N 128
#define H_N 512
#define G4H 2048   // 4*H

// ---------------------------------------------------------------- helpers ---
__device__ __forceinline__ v8f zero8() {
  v8f z = {0.f,0.f,0.f,0.f,0.f,0.f,0.f,0.f};
  return z;
}

// A-matrix 16x32 bf16 fragment (row-major source, row = lane&15):
// halves 0..7  = K[k0+8g .. k0+8g+7], halves 8..15 = K[k0+16+8g .. +7]
__device__ __forceinline__ v16bf frag_a(const bf16* row, int k0, int g) {
  v8bf lo = *(const v8bf*)(row + k0 + 8 * g);
  v8bf hi = *(const v8bf*)(row + k0 + 16 + 8 * g);
  return __builtin_shufflevector(lo, hi, 0,1,2,3,4,5,6,7,8,9,10,11,12,13,14,15);
}
// B-matrix 32x16 bf16 fragment (Bt row-major N x K, row = n = lane&15):
// halves 0..7  = K[k0+8g .. k0+8g+7], halves 8..15 = K[k0+16+8g .. +7]
__device__ __forceinline__ v16bf frag_b(const bf16* row, int k0, int g) {
  v8bf lo = *(const v8bf*)(row + k0 + 8 * g);
  v8bf hi = *(const v8bf*)(row + k0 + 16 + 8 * g);
  return __builtin_shufflevector(lo, hi, 0,1,2,3,4,5,6,7,8,9,10,11,12,13,14,15);
}
__device__ __forceinline__ v8f wmma_bf(v16bf a, v16bf b, v8f c) {
  return __builtin_amdgcn_wmma_f32_16x16x32_bf16(false, a, false, b, (short)0, c,
                                                 false, false);
}
__device__ __forceinline__ float sigf(float x) { return 1.f / (1.f + __expf(-x)); }

// ------------------------------------------------------------ pack kernels ---
__global__ void cvt_bf16_kernel(const float* __restrict__ in, bf16* __restrict__ out,
                                int n) {
  for (int i = blockIdx.x * blockDim.x + threadIdx.x; i < n;
       i += gridDim.x * blockDim.x)
    out[i] = (bf16)in[i];
}

// wcat[n][k] = k<512 ? w_ih[n][k] : w_hh[n][k-512]   (2048 x 1024)
__global__ void build_wcat_kernel(const float* __restrict__ wih,
                                  const float* __restrict__ whh,
                                  bf16* __restrict__ wcat) {
  const int n = G4H * 1024;
  for (int i = blockIdx.x * blockDim.x + threadIdx.x; i < n;
       i += gridDim.x * blockDim.x) {
    int k = i & 1023, r = i >> 10;
    float v = (k < H_N) ? wih[(size_t)r * H_N + k] : whh[(size_t)r * H_N + (k - H_N)];
    wcat[i] = (bf16)v;
  }
}

__global__ void embed_gather_kernel(const int* __restrict__ tok,
                                    const float* __restrict__ embed,
                                    bf16* __restrict__ xb) {
  const int n = T_N * B_N * H_N;
  for (int i = blockIdx.x * blockDim.x + threadIdx.x; i < n;
       i += gridDim.x * blockDim.x) {
    int e = i & (H_N - 1), r = i >> 9;
    xb[i] = (bf16)embed[(size_t)tok[r] * H_N + e];
  }
}

// mem (S,B,D) f32 -> o_sd (B,S,D) bf16 and o_ds (B,D,S) bf16
__global__ void pack_mem_kernel(const float* __restrict__ mem, bf16* __restrict__ o_sd,
                                bf16* __restrict__ o_ds, int S) {
  const int D = H_N, B = B_N;
  int n = S * B * D;
  for (int i = blockIdx.x * blockDim.x + threadIdx.x; i < n;
       i += gridDim.x * blockDim.x) {
    int d = i & (D - 1);
    int r = i >> 9;
    int b = r & (B - 1);
    int s = r >> 6;
    float v = mem[i];
    o_sd[((size_t)b * S + s) * D + d] = (bf16)v;
    o_ds[((size_t)b * D + d) * S + s] = (bf16)v;
  }
}

__global__ void init_state_kernel(bf16* __restrict__ h0, float* __restrict__ c) {
  const int n = B_N * H_N;
  for (int i = blockIdx.x * blockDim.x + threadIdx.x; i < n;
       i += gridDim.x * blockDim.x) {
    h0[i] = (bf16)0.f;
    c[i] = 0.f;
  }
}

__global__ void copy_f32_kernel(const float* __restrict__ in, float* __restrict__ out,
                                int n) {
  for (int i = blockIdx.x * blockDim.x + threadIdx.x; i < n;
       i += gridDim.x * blockDim.x)
    out[i] = in[i];
}

// ------------------------------------------------------------- LSTM step ----
// gates = [x_t ; h_prev](64x1024) @ wcat^T(2048x1024) + b_ih + b_hh, then cell.
// grid: 8 blocks (64 h-cols each), 256 threads (8 waves, 8 WMMA tiles/wave).
__global__ __launch_bounds__(256) void lstm_step_kernel(
    const bf16* __restrict__ x_bf, const bf16* __restrict__ wcat,
    const float* __restrict__ b_ih, const float* __restrict__ b_hh,
    const bf16* __restrict__ h_prev, bf16* __restrict__ h_next,
    float* __restrict__ c, bf16* __restrict__ outs_bf, float* __restrict__ hlast,
    int t) {
  const int H = H_N, B = B_N;
  int tid = threadIdx.x;
  int lane = tid & 31, lo = lane & 15, g = lane >> 4;
  int wave = tid >> 5;
  int mt = wave >> 1, half = wave & 1;
  int nb = blockIdx.x;  // 0..7
  int arow = mt * 16 + lo;

  const bf16* xrow = x_bf + ((size_t)(t * B + arow)) * H;
  const bf16* hrow = h_prev + (size_t)arow * H;

  const bf16* brow[4][2];
  float biasv[4][2];
  v8f acc[4][2];
#pragma unroll
  for (int ch = 0; ch < 4; ++ch)
#pragma unroll
    for (int ctl = 0; ctl < 2; ++ctl) {
      int n = ch * H + nb * 64 + (2 * half + ctl) * 16 + lo;
      brow[ch][ctl] = wcat + (size_t)n * 1024;
      biasv[ch][ctl] = b_ih[n] + b_hh[n];
      acc[ch][ctl] = zero8();
    }

  for (int k0 = 0; k0 < 1024; k0 += 32) {
    v16bf af = (k0 < H) ? frag_a(xrow, k0, g) : frag_a(hrow, k0 - H, g);
#pragma unroll
    for (int ch = 0; ch < 4; ++ch)
#pragma unroll
      for (int ctl = 0; ctl < 2; ++ctl)
        acc[ch][ctl] = wmma_bf(af, frag_b(brow[ch][ctl], k0, g), acc[ch][ctl]);
  }

#pragma unroll
  for (int ctl = 0; ctl < 2; ++ctl) {
    int j = nb * 64 + (2 * half + ctl) * 16 + lo;
#pragma unroll
    for (int r = 0; r < 8; ++r) {
      int b = mt * 16 + r + 8 * g;
      float iv = acc[0][ctl][r] + biasv[0][ctl];
      float fv = acc[1][ctl][r] + biasv[1][ctl];
      float gv = acc[2][ctl][r] + biasv[2][ctl];
      float ov = acc[3][ctl][r] + biasv[3][ctl];
      float cn = sigf(fv) * c[b * H + j] + sigf(iv) * tanhf(gv);
      float hn = sigf(ov) * tanhf(cn);
      c[b * H + j] = cn;
      h_next[b * H + j] = (bf16)hn;
      outs_bf[((size_t)(t * B + b)) * H + j] = (bf16)hn;
      if (hlast) hlast[b * H + j] = hn;
    }
  }
}

// ---------------------------------------------------------- generic GEMM ----
// C[m,n] = act( sum_k A[m,k]*Bt[n,k] + bias[n] ), A = concat(A1,A2) at ksplit.
// act: 0=none, 1=tanh, 2=sigmoid-gate blend: out = s*blend1 + (1-s)*blend2.
// One wave per 16x64 slab (1 M-tile x 4 N-tiles, A fragment reused 4x):
// grid = (N/64, M/16), block = 32.
__global__ __launch_bounds__(32) void gemm_bf16_kernel(
    const bf16* __restrict__ A1, const bf16* __restrict__ A2, int ksplit, int lda1,
    int lda2, const bf16* __restrict__ Bt, int ldb, const float* __restrict__ bias,
    float* __restrict__ outF, bf16* __restrict__ outB, int ldc, int Kdim, int act,
    const float* __restrict__ blend1, const float* __restrict__ blend2) {
  int lane = threadIdx.x, lo = lane & 15, g = lane >> 4;
  int m0 = blockIdx.y * 16, n0 = blockIdx.x * 64;
  const bf16* a1 = A1 + (size_t)(m0 + lo) * lda1;
  const bf16* a2 = A2 ? A2 + (size_t)(m0 + lo) * lda2 : A1;
  const bf16* br[4];
  v8f acc[4];
#pragma unroll
  for (int j = 0; j < 4; ++j) {
    br[j] = Bt + (size_t)(n0 + j * 16 + lo) * ldb;
    acc[j] = zero8();
  }

  for (int k0 = 0; k0 < Kdim; k0 += 32) {
    v16bf af = (k0 < ksplit) ? frag_a(a1, k0, g) : frag_a(a2, k0 - ksplit, g);
#pragma unroll
    for (int j = 0; j < 4; ++j)
      acc[j] = wmma_bf(af, frag_b(br[j], k0, g), acc[j]);
  }

#pragma unroll
  for (int j = 0; j < 4; ++j) {
    int n = n0 + j * 16 + lo;
    float bs = bias ? bias[n] : 0.f;
#pragma unroll
    for (int r = 0; r < 8; ++r) {
      int m = m0 + r + 8 * g;
      float v = acc[j][r] + bs;
      if (act == 1) v = tanhf(v);
      size_t oi = (size_t)m * ldc + n;
      if (act == 2) {
        float s = sigf(v);
        v = s * blend1[oi] + (1.f - s) * blend2[oi];
      }
      if (outF) outF[oi] = v;
      if (outB) outB[oi] = (bf16)v;
    }
  }
}

// -------------------------------------------------------- fused attention ---
// Per (16 t-rows, batch b): scores = qp @ mem_sd^T, mask, softmax over S,
// ctx = wts @ mem_ds^T. grid = (T/16, B), block = 256.
__global__ __launch_bounds__(256) void attn_fused_kernel(
    const bf16* __restrict__ qp,                 // (T*B, 512)
    const bf16* __restrict__ mem_sd,             // (B, S, 512)
    const bf16* __restrict__ mem_ds,             // (B, 512, S)
    const unsigned char* __restrict__ mask,      // (B, S), nonzero = pad
    float* __restrict__ ctxF, bf16* __restrict__ ctxB,  // (T*B, 512)
    float* __restrict__ wts_out,                 // (T, B, S) or null
    int S) {
  const int D = H_N, B = B_N;
  __shared__ __align__(16) float sc[16 * 256];
  __shared__ __align__(16) bf16 swt[16 * 256];
  __shared__ float redA[16 * 16];
  __shared__ float redB[16 * 16];

  int tid = threadIdx.x, lane = tid & 31, lo = lane & 15, g = lane >> 4;
  int wave = tid >> 5;
  int t0 = blockIdx.x * 16, b = blockIdx.y;
  const bf16* qrow = qp + ((size_t)((t0 + lo) * B + b)) * D;
  int ntiles = S >> 4;
  bool two = (ntiles == 16);  // S==256 -> 2 score tiles per wave, else 1

  // phase 1: score tiles (A fragment reused across the wave's tiles)
  {
    int scol0 = wave * 16 + lo;
    int scol1 = two ? (wave + 8) * 16 + lo : scol0;
    const bf16* mrow0 = mem_sd + ((size_t)b * S + scol0) * D;
    const bf16* mrow1 = mem_sd + ((size_t)b * S + scol1) * D;
    v8f a0 = zero8(), a1 = zero8();
    for (int k0 = 0; k0 < D; k0 += 32) {
      v16bf af = frag_a(qrow, k0, g);
      a0 = wmma_bf(af, frag_b(mrow0, k0, g), a0);
      if (two) a1 = wmma_bf(af, frag_b(mrow1, k0, g), a1);
    }
    bool m0 = mask[b * S + scol0] != 0;
#pragma unroll
    for (int r = 0; r < 8; ++r)
      sc[(r + 8 * g) * 256 + scol0] = m0 ? -1e9f : a0[r];
    if (two) {
      bool m1 = mask[b * S + scol1] != 0;
#pragma unroll
      for (int r = 0; r < 8; ++r)
        sc[(r + 8 * g) * 256 + scol1] = m1 ? -1e9f : a1[r];
    }
  }
  __syncthreads();

  // phase 2: softmax (16 threads per row)
  int row = tid >> 4, c16 = tid & 15;
  float mx = -3.4e38f;
  for (int k = 0; k < ntiles; ++k) mx = fmaxf(mx, sc[row * 256 + c16 + 16 * k]);
  redA[row * 16 + c16] = mx;
  __syncthreads();
  float rmax = -3.4e38f;
  for (int i = 0; i < 16; ++i) rmax = fmaxf(rmax, redA[row * 16 + i]);
  float lsum = 0.f;
  for (int k = 0; k < ntiles; ++k) {
    int idx = row * 256 + c16 + 16 * k;
    float e = __expf(sc[idx] - rmax);
    sc[idx] = e;
    lsum += e;
  }
  redB[row * 16 + c16] = lsum;
  __syncthreads();
  float rsum = 0.f;
  for (int i = 0; i < 16; ++i) rsum += redB[row * 16 + i];
  float inv = 1.f / rsum;
  for (int k = 0; k < ntiles; ++k) {
    int col = c16 + 16 * k;
    float w = sc[row * 256 + col] * inv;
    swt[row * 256 + col] = (bf16)w;
    if (wts_out) wts_out[((size_t)((t0 + row) * B + b)) * S + col] = w;
  }
  __syncthreads();

  // phase 3: context tiles (N = 512 -> 4 tiles per wave, A fragment reused 4x)
  {
    const bf16* arow = swt + (size_t)lo * 256;
    const bf16* mrow[4];
    v8f a[4];
#pragma unroll
    for (int i = 0; i < 4; ++i) {
      int n = (wave * 4 + i) * 16 + lo;
      mrow[i] = mem_ds + ((size_t)b * D + n) * S;
      a[i] = zero8();
    }
    for (int k0 = 0; k0 < S; k0 += 32) {
      v16bf af = frag_a(arow, k0, g);
#pragma unroll
      for (int i = 0; i < 4; ++i)
        a[i] = wmma_bf(af, frag_b(mrow[i], k0, g), a[i]);
    }
#pragma unroll
    for (int i = 0; i < 4; ++i) {
      int n = (wave * 4 + i) * 16 + lo;
#pragma unroll
      for (int r = 0; r < 8; ++r) {
        size_t oi = ((size_t)((t0 + r + 8 * g) * B + b)) * D + n;
        if (ctxF) ctxF[oi] = a[i][r];
        ctxB[oi] = (bf16)a[i][r];
      }
    }
  }
}

// ------------------------------------------------------------------ host ----
extern "C" void kernel_launch(void* const* d_in, const int* in_sizes, int n_in,
                              void* d_out, int out_size, void* d_ws, size_t ws_size,
                              hipStream_t stream) {
  (void)in_sizes; (void)n_in; (void)out_size;
  const int* tok = (const int*)d_in[0];
  const float* mem_src = (const float*)d_in[1];
  const unsigned char* msk_src = (const unsigned char*)d_in[2];
  const float* mem_ske = (const float*)d_in[3];
  const unsigned char* msk_ske = (const unsigned char*)d_in[4];
  const float* embed = (const float*)d_in[5];
  const float* w_ih = (const float*)d_in[6];
  const float* w_hh = (const float*)d_in[7];
  const float* b_ih = (const float*)d_in[8];
  const float* b_hh = (const float*)d_in[9];
  const float* w_attn_src = (const float*)d_in[10];
  const float* w_proj = (const float*)d_in[11];
  const float* b_proj = (const float*)d_in[12];
  const float* w_attn_ske = (const float*)d_in[13];
  const float* w_gate = (const float*)d_in[14];
  const float* b_gate = (const float*)d_in[15];

  const size_t TB = (size_t)T_N * B_N;  // 8192

  char* ws = (char*)d_ws;
  size_t off = 0;
  auto carve = [&](size_t bytes) -> void* {
    void* p = ws + off;
    off = (off + bytes + 255) & ~(size_t)255;
    return p;
  };
  bf16* wcat_bf   = (bf16*)carve((size_t)G4H * 1024 * 2);
  bf16* was_bf    = (bf16*)carve((size_t)H_N * H_N * 2);
  bf16* wproj_bf  = (bf16*)carve((size_t)H_N * 1024 * 2);
  bf16* wask_bf   = (bf16*)carve((size_t)H_N * H_N * 2);
  bf16* wgate_bf  = (bf16*)carve((size_t)H_N * 1024 * 2);
  bf16* x_bf      = (bf16*)carve(TB * H_N * 2);
  bf16* h_bf      = (bf16*)carve((size_t)2 * B_N * H_N * 2);
  float* c_buf    = (float*)carve((size_t)B_N * H_N * 4);
  bf16* outs_bf   = (bf16*)carve(TB * H_N * 2);
  bf16* msrc_sd   = (bf16*)carve((size_t)B_N * S_N * H_N * 2);
  bf16* msrc_ds   = (bf16*)carve((size_t)B_N * H_N * S_N * 2);
  bf16* mske_sd   = (bf16*)carve((size_t)B_N * K_N * H_N * 2);
  bf16* mske_ds   = (bf16*)carve((size_t)B_N * H_N * K_N * 2);
  bf16* qp_bf     = (bf16*)carve(TB * H_N * 2);
  bf16* asrc_bf   = (bf16*)carve(TB * H_N * 2);
  float* wsrc_f   = (float*)carve(TB * H_N * 4);
  bf16* wsrc_bf   = (bf16*)carve(TB * H_N * 2);
  float* aske_f   = (float*)carve(TB * H_N * 4);
  bf16* aske_bf   = (bf16*)carve(TB * H_N * 2);
  if (off > ws_size) return;

  float* fout = (float*)d_out;
  float* out_final = fout;                                       // T*B*H
  float* out_score = fout + TB * H_N;                            // T*B*K
  float* out_h = out_score + TB * K_N;                           // B*H
  float* out_c = out_h + (size_t)B_N * H_N;                      // B*H

  // pack phase
  build_wcat_kernel<<<1024, 256, 0, stream>>>(w_ih, w_hh, wcat_bf);
  cvt_bf16_kernel<<<512, 256, 0, stream>>>(w_attn_src, was_bf, H_N * H_N);
  cvt_bf16_kernel<<<512, 256, 0, stream>>>(w_proj, wproj_bf, H_N * 1024);
  cvt_bf16_kernel<<<512, 256, 0, stream>>>(w_attn_ske, wask_bf, H_N * H_N);
  cvt_bf16_kernel<<<512, 256, 0, stream>>>(w_gate, wgate_bf, H_N * 1024);
  embed_gather_kernel<<<2048, 256, 0, stream>>>(tok, embed, x_bf);
  pack_mem_kernel<<<2048, 256, 0, stream>>>(mem_src, msrc_sd, msrc_ds, S_N);
  pack_mem_kernel<<<2048, 256, 0, stream>>>(mem_ske, mske_sd, mske_ds, K_N);
  init_state_kernel<<<64, 256, 0, stream>>>(h_bf, c_buf);

  // recurrent LSTM: 128 stream-ordered steps, double-buffered h
  for (int t = 0; t < T_N; ++t) {
    const bf16* hp = h_bf + (size_t)(t & 1) * B_N * H_N;
    bf16* hn = h_bf + (size_t)((t + 1) & 1) * B_N * H_N;
    lstm_step_kernel<<<8, 256, 0, stream>>>(x_bf, wcat_bf, b_ih, b_hh, hp, hn, c_buf,
                                            outs_bf, (t == T_N - 1) ? out_h : nullptr,
                                            t);
  }
  copy_f32_kernel<<<64, 256, 0, stream>>>(c_buf, out_c, B_N * H_N);

  dim3 g8x512(H_N / 64, TB / 16);  // (8, 512): 16x64 slab per wave

  // qp_src = outs @ w_attn_src^T  (bf16 out)
  gemm_bf16_kernel<<<g8x512, 32, 0, stream>>>(outs_bf, nullptr, H_N, H_N, 0, was_bf,
                                              H_N, nullptr, nullptr, qp_bf, H_N, H_N,
                                              0, nullptr, nullptr);
  // src attention -> asrc_bf
  attn_fused_kernel<<<dim3(T_N / 16, B_N), 256, 0, stream>>>(
      qp_bf, msrc_sd, msrc_ds, msk_src, nullptr, asrc_bf, nullptr, S_N);
  // with_src = tanh([outs ; attn_src] @ w_proj^T + b_proj)
  gemm_bf16_kernel<<<g8x512, 32, 0, stream>>>(outs_bf, asrc_bf, H_N, H_N, H_N,
                                              wproj_bf, 1024, b_proj, wsrc_f, wsrc_bf,
                                              H_N, 1024, 1, nullptr, nullptr);
  // qp_ske = with_src @ w_attn_ske^T  (reuse qp_bf)
  gemm_bf16_kernel<<<g8x512, 32, 0, stream>>>(wsrc_bf, nullptr, H_N, H_N, 0, wask_bf,
                                              H_N, nullptr, nullptr, qp_bf, H_N, H_N,
                                              0, nullptr, nullptr);
  // ske attention -> aske (f32+bf16), weights straight into d_out
  attn_fused_kernel<<<dim3(T_N / 16, B_N), 256, 0, stream>>>(
      qp_bf, mske_sd, mske_ds, msk_ske, aske_f, aske_bf, out_score, K_N);
  // gate + blend: final = sig(z)*attn_ske + (1-sig(z))*with_src
  gemm_bf16_kernel<<<g8x512, 32, 0, stream>>>(wsrc_bf, aske_bf, H_N, H_N, H_N,
                                              wgate_bf, 1024, b_gate, out_final,
                                              nullptr, H_N, 1024, 2, aske_f, wsrc_f);
}
